// AgentCapabilityEstimator_57260503990878
// MI455X (gfx1250) — matrix operand order repaired and checked by reading, not compiled, not run on hardware
//
#include <hip/hip_runtime.h>
#include <stdint.h>

typedef __attribute__((ext_vector_type(16))) __bf16 v16bf;
typedef __attribute__((ext_vector_type(8)))  __bf16 v8bf;
typedef __attribute__((ext_vector_type(8)))  float  v8f;

#define B_ROWS 16384
#define OBS_D  512
#define H_D    1024
#define N_AG   9
#define LDS_STRIDE 80          // 64B of A data per row + 16B pad
#define KSUB   32              // K per WMMA step
#define NSUB   4               // K-substeps per super-stage
#define KSUPER (KSUB * NSUB)   // 128
#define TILE_B (64 * LDS_STRIDE)   // 5120 B per 64x32 sub-tile
#define BUF_B  (NSUB * TILE_B)     // 20480 B per super-buffer

#define WMMA_BF16(af, bf, c) \
  __builtin_amdgcn_wmma_f32_16x16x32_bf16(false, (af), false, (bf), (short)0, (c), false, false)

// ---------------------------------------------------------------------------
// Prep kernels
// ---------------------------------------------------------------------------
__global__ void cvt_f32_to_bf16(const float* __restrict__ src,
                                __bf16* __restrict__ dst, long long n) {
  long long i = ((long long)blockIdx.x * blockDim.x + threadIdx.x) * 4;
  if (i + 4 <= n) {
    float4 v = *(const float4*)(src + i);
    dst[i + 0] = (__bf16)v.x;
    dst[i + 1] = (__bf16)v.y;
    dst[i + 2] = (__bf16)v.z;
    dst[i + 3] = (__bf16)v.w;
  } else {
    for (; i < n; ++i) dst[i] = (__bf16)src[i];
  }
}

// src is [K][N] row-major (jax weight); dst is [N][K] row-major bf16
__global__ void transpose_cvt(const float* __restrict__ src,
                              __bf16* __restrict__ dst, int K, int N) {
  long long idx = (long long)blockIdx.x * blockDim.x + threadIdx.x;
  if (idx >= (long long)K * N) return;
  int n = (int)(idx / K);
  int k = (int)(idx % K);
  dst[idx] = (__bf16)src[(long long)k * N + n];
}

// src is [2K][N]; dst[n*K+k] = bf16(src[k][n] + src[k+K][n])  (concat(g,g) fold)
__global__ void transpose_fold_cvt(const float* __restrict__ src,
                                   __bf16* __restrict__ dst, int K, int N) {
  long long idx = (long long)blockIdx.x * blockDim.x + threadIdx.x;
  if (idx >= (long long)K * N) return;
  int n = (int)(idx / K);
  int k = (int)(idx % K);
  float v = src[(long long)k * N + n] + src[(long long)(k + K) * N + n];
  dst[idx] = (__bf16)v;
}

// Concatenate head output vectors and head hidden biases: [cov | trk | coop]
__global__ void build_head_vecs(const float* __restrict__ Wc2,
                                const float* __restrict__ Wt2,
                                const float* __restrict__ Wk2,
                                const float* __restrict__ bc1,
                                const float* __restrict__ bt1,
                                const float* __restrict__ bk1,
                                float* __restrict__ w2cat,
                                float* __restrict__ bcat) {
  int i = blockIdx.x * blockDim.x + threadIdx.x;  // 0..2047
  if (i >= 2048) return;
  float w, b;
  if (i < 512)       { w = Wc2[i];        b = bc1[i]; }
  else if (i < 1024) { w = Wt2[i - 512];  b = bt1[i - 512]; }
  else               { w = Wk2[i - 1024]; b = bk1[i - 1024]; }
  w2cat[i] = w;
  bcat[i]  = b;
}

// ---------------------------------------------------------------------------
// Shared WMMA GEMM core.
// Wave computes 32(M) x 64(N); block = 8 waves (2 waveM x 4 waveN) = 64 x 256.
// A staged as double-buffered 64x128 super-tiles (4 sub-tiles of 64x32) via
// global_load_async_to_lds_b128 -> only 2 barriers per 128 K-elements.
// B fragments software-pipelined in registers across the barrier-free region.
// ---------------------------------------------------------------------------
__device__ __forceinline__ void issue_async_tile(const __bf16* gsrcp, uint32_t ldst) {
  uint64_t gsrc = (uint64_t)(size_t)gsrcp;
  asm volatile("global_load_async_to_lds_b128 %0, %1, off"
               :: "v"(ldst), "v"(gsrc) : "memory");
}

__device__ __forceinline__ v16bf load_afrag(const char* sb, int lrow, int half) {
  v8bf alo = *(const v8bf*)(sb + lrow * LDS_STRIDE + half * 32);
  v8bf ahi = *(const v8bf*)(sb + lrow * LDS_STRIDE + half * 32 + 16);
  return __builtin_shufflevector(alo, ahi,
      0,1,2,3,4,5,6,7,8,9,10,11,12,13,14,15);
}

template<int K>
__device__ __forceinline__ void gemm_core(const __bf16* __restrict__ A,
                                          const __bf16* __restrict__ Wt,
                                          char* smem,       // [2 * BUF_B]
                                          v8f (&acc)[2][4]) {
  const int tid  = threadIdx.x;
  const int lane = tid & 31;
  const int wave = tid >> 5;
  const int l16  = lane & 15;
  const int half = lane >> 4;
  const int waveM = wave & 1;
  const int waveN = wave >> 1;
  const int blockRow = blockIdx.y * 64;
  const int colBase  = blockIdx.x * 256 + waveN * 64;

  // Each of the 256 threads owns one 16B chunk of a 64x32 sub-tile per K step.
  const int cRow   = tid >> 2;   // 0..63
  const int cChunk = tid & 3;    // 0..3
  const __bf16* gA0 = A + (size_t)(blockRow + cRow) * K + cChunk * 8;
  const uint32_t ldsBase =
      (uint32_t)(size_t)smem + (uint32_t)(cRow * LDS_STRIDE + cChunk * 16);

  const __bf16* bp[4];
#pragma unroll
  for (int j = 0; j < 4; ++j)
    bp[j] = Wt + (size_t)(colBase + j * 16 + l16) * K + half * 16;

#pragma unroll
  for (int mt = 0; mt < 2; ++mt)
#pragma unroll
    for (int j = 0; j < 4; ++j)
      acc[mt][j] = v8f{0.f,0.f,0.f,0.f,0.f,0.f,0.f,0.f};

  const int lrow0 = waveM * 32 + l16;
  const int lrow1 = waveM * 32 + 16 + l16;

  // Prologue: super-tile 0 into buffer 0; first B fragments.
#pragma unroll
  for (int s = 0; s < NSUB; ++s)
    issue_async_tile(gA0 + s * KSUB, ldsBase + s * TILE_B);

  v16bf bcur[4];
#pragma unroll
  for (int j = 0; j < 4; ++j) bcur[j] = *(const v16bf*)(bp[j]);

  int bufSel = 0;
  int k0 = 0;
  // Main super-stages (all but last): prefetch next super-tile unconditionally.
  for (; k0 + KSUPER < K; k0 += KSUPER, bufSel ^= 1) {
#pragma unroll
    for (int s = 0; s < NSUB; ++s)
      issue_async_tile(gA0 + k0 + KSUPER + s * KSUB,
                       ldsBase + (bufSel ^ 1) * BUF_B + s * TILE_B);
    asm volatile("s_wait_asynccnt 0x4" ::: "memory");  // current buffer ready
    __syncthreads();
    const char* base = smem + bufSel * BUF_B;
#pragma unroll
    for (int s = 0; s < NSUB; ++s) {
      v16bf bnext[4];
#pragma unroll
      for (int j = 0; j < 4; ++j)
        bnext[j] = *(const v16bf*)(bp[j] + k0 + s * KSUB + KSUB);
      v16bf af0 = load_afrag(base + s * TILE_B, lrow0, half);
      v16bf af1 = load_afrag(base + s * TILE_B, lrow1, half);
#pragma unroll
      for (int j = 0; j < 4; ++j) acc[0][j] = WMMA_BF16(af0, bcur[j], acc[0][j]);
#pragma unroll
      for (int j = 0; j < 4; ++j) acc[1][j] = WMMA_BF16(af1, bcur[j], acc[1][j]);
#pragma unroll
      for (int j = 0; j < 4; ++j) bcur[j] = bnext[j];
    }
    __syncthreads();
  }

  // Tail super-stage (k0 == K - KSUPER): nothing left to prefetch.
  asm volatile("s_wait_asynccnt 0x0" ::: "memory");
  __syncthreads();
  const char* base = smem + bufSel * BUF_B;
#pragma unroll
  for (int s = 0; s < NSUB; ++s) {
    v16bf bnext[4];
    if (s < NSUB - 1) {
#pragma unroll
      for (int j = 0; j < 4; ++j)
        bnext[j] = *(const v16bf*)(bp[j] + k0 + s * KSUB + KSUB);
    }
    v16bf af0 = load_afrag(base + s * TILE_B, lrow0, half);
    v16bf af1 = load_afrag(base + s * TILE_B, lrow1, half);
#pragma unroll
    for (int j = 0; j < 4; ++j) acc[0][j] = WMMA_BF16(af0, bcur[j], acc[0][j]);
#pragma unroll
    for (int j = 0; j < 4; ++j) acc[1][j] = WMMA_BF16(af1, bcur[j], acc[1][j]);
    if (s < NSUB - 1) {
#pragma unroll
      for (int j = 0; j < 4; ++j) bcur[j] = bnext[j];
    }
  }
}

// ---------------------------------------------------------------------------
// Trunk GEMM: out[m][n] = relu(A @ Wt^T + bias), bf16 store.
// ---------------------------------------------------------------------------
template<int K, int Nout>
__global__ __launch_bounds__(256)
void gemm_relu_store_bf16(const __bf16* __restrict__ A,
                          const __bf16* __restrict__ Wt,
                          const float* __restrict__ bias,
                          __bf16* __restrict__ out) {
  __shared__ __align__(16) char smem[2 * BUF_B];
  v8f acc[2][4];
  gemm_core<K>(A, Wt, smem, acc);

  const int lane = threadIdx.x & 31;
  const int wave = threadIdx.x >> 5;
  const int l16  = lane & 15;
  const int half = lane >> 4;
  const int waveM = wave & 1;
  const int waveN = wave >> 1;
  const int blockRow = blockIdx.y * 64;
  const int colBase  = blockIdx.x * 256 + waveN * 64;

#pragma unroll
  for (int mt = 0; mt < 2; ++mt) {
    const int rowBase = blockRow + waveM * 32 + mt * 16;
#pragma unroll
    for (int j = 0; j < 4; ++j) {
      const int colG = colBase + j * 16 + l16;
      const float bv = bias[colG];
      size_t ob = (size_t)(rowBase + half * 8) * Nout + colG;
#pragma unroll
      for (int r = 0; r < 8; ++r) {
        float v = acc[mt][j][r] + bv;
        v = v > 0.f ? v : 0.f;
        out[ob + (size_t)r * Nout] = (__bf16)v;
      }
    }
  }
}

// ---------------------------------------------------------------------------
// Heads GEMM: h = relu(g2 @ Wht^T + bcat), fused dot with w2cat.
// Epilogue reduces across the 16 column lanes and writes deterministic
// partials part[row][subtile] (128 subtiles: 32 cov | 32 trk | 64 coop).
// ---------------------------------------------------------------------------
template<int K>
__global__ __launch_bounds__(256)
void heads_gemm_partial(const __bf16* __restrict__ A,
                        const __bf16* __restrict__ Wt,
                        const float* __restrict__ bias,
                        const float* __restrict__ w2,
                        float* __restrict__ part) {
  __shared__ __align__(16) char smem[2 * BUF_B];
  v8f acc[2][4];
  gemm_core<K>(A, Wt, smem, acc);

  const int lane = threadIdx.x & 31;
  const int wave = threadIdx.x >> 5;
  const int l16  = lane & 15;
  const int half = lane >> 4;
  const int waveM = wave & 1;
  const int waveN = wave >> 1;
  const int blockRow = blockIdx.y * 64;
  const int colBase  = blockIdx.x * 256 + waveN * 64;

  const int subBase = (colBase >> 4);  // global 16-col subtile index of j=0
#pragma unroll
  for (int mt = 0; mt < 2; ++mt) {
    const int rowBase = blockRow + waveM * 32 + mt * 16;
#pragma unroll
    for (int j = 0; j < 4; ++j) {
      const int colG = colBase + j * 16 + l16;
      const float bv = bias[colG];
      const float wv = w2[colG];
#pragma unroll
      for (int r = 0; r < 8; ++r) {
        float v = acc[mt][j][r] + bv;
        v = (v > 0.f ? v : 0.f) * wv;
        // reduce across the 16 column lanes (stays inside each half)
        v += __shfl_xor(v, 1, 32);
        v += __shfl_xor(v, 2, 32);
        v += __shfl_xor(v, 4, 32);
        v += __shfl_xor(v, 8, 32);
        if (l16 == 0) {
          int rowG = rowBase + r + 8 * half;
          part[(size_t)rowG * 128 + subBase + j] = v;
        }
      }
    }
  }
}

// ---------------------------------------------------------------------------
// Finalize: sum partials per head, sigmoid, broadcast to [B,9] x 3 heads.
// ---------------------------------------------------------------------------
__global__ void finalize_kernel(const float* __restrict__ part,
                                const float* __restrict__ bc2,
                                const float* __restrict__ bt2,
                                const float* __restrict__ bk2,
                                float* __restrict__ out) {
  int b = blockIdx.x * blockDim.x + threadIdx.x;
  if (b >= B_ROWS) return;
  const float* p = part + (size_t)b * 128;
  float sc = 0.f, st = 0.f, sk = 0.f;
  for (int i = 0;  i < 32;  ++i) sc += p[i];
  for (int i = 32; i < 64;  ++i) st += p[i];
  for (int i = 64; i < 128; ++i) sk += p[i];
  sc = 1.f / (1.f + __expf(-(sc + bc2[0])));
  st = 1.f / (1.f + __expf(-(st + bt2[0])));
  sk = 1.f / (1.f + __expf(-(sk + bk2[0])));
  size_t o = (size_t)b * N_AG;
  const size_t plane = (size_t)B_ROWS * N_AG;
#pragma unroll
  for (int j = 0; j < N_AG; ++j) {
    out[o + j]             = sc;
    out[plane + o + j]     = st;
    out[2 * plane + o + j] = sk;
  }
}

// ---------------------------------------------------------------------------
// Launch
// ---------------------------------------------------------------------------
extern "C" void kernel_launch(void* const* d_in, const int* in_sizes, int n_in,
                              void* d_out, int out_size, void* d_ws, size_t ws_size,
                              hipStream_t stream) {
  const float* obs = (const float*)d_in[0];
  // d_in[1] = agent_positions (unused by the reference math)
  const float* W1  = (const float*)d_in[2];
  const float* b1  = (const float*)d_in[3];
  const float* W2  = (const float*)d_in[4];
  const float* b2  = (const float*)d_in[5];
  const float* Wc1 = (const float*)d_in[6];
  const float* bc1 = (const float*)d_in[7];
  const float* Wc2 = (const float*)d_in[8];
  const float* bc2 = (const float*)d_in[9];
  const float* Wt1 = (const float*)d_in[10];
  const float* bt1 = (const float*)d_in[11];
  const float* Wt2 = (const float*)d_in[12];
  const float* bt2 = (const float*)d_in[13];
  const float* Wk1 = (const float*)d_in[14];
  const float* bk1 = (const float*)d_in[15];
  const float* Wk2 = (const float*)d_in[16];
  const float* bk2 = (const float*)d_in[17];

  char* ws = (char*)d_ws;
  __bf16* obs_bf = (__bf16*)(ws + 0LL);          // 16384*512*2   = 16 MiB
  __bf16* g1     = (__bf16*)(ws + 16777216LL);   // 16384*1024*2  = 32 MiB
  __bf16* g2     = (__bf16*)(ws + 50331648LL);   // 32 MiB
  __bf16* W1t    = (__bf16*)(ws + 83886080LL);   // 1024*512*2    = 1 MiB
  __bf16* W2t    = (__bf16*)(ws + 84934656LL);   // 1024*1024*2   = 2 MiB
  __bf16* Wht    = (__bf16*)(ws + 87031808LL);   // 2048*1024*2   = 4 MiB
  float*  w2cat  = (float*)(ws + 91226112LL);    // 2048*4
  float*  bcat   = (float*)(ws + 91234304LL);    // 2048*4
  float*  part   = (float*)(ws + 91242496LL);    // 16384*128*4   = 8 MiB

  // --- prep ---
  cvt_f32_to_bf16<<<8192, 256, 0, stream>>>(obs, obs_bf, (long long)B_ROWS * OBS_D);
  transpose_cvt<<<2048, 256, 0, stream>>>(W1, W1t, OBS_D, H_D);                  // [512,1024] -> [1024][512]
  transpose_cvt<<<4096, 256, 0, stream>>>(W2, W2t, H_D, H_D);                    // [1024,1024] -> [1024][1024]
  transpose_cvt<<<2048, 256, 0, stream>>>(Wc1, Wht, H_D, H_D / 2);               // rows 0..511
  transpose_cvt<<<2048, 256, 0, stream>>>(Wt1, Wht + 512 * H_D, H_D, H_D / 2);   // rows 512..1023
  transpose_fold_cvt<<<4096, 256, 0, stream>>>(Wk1, Wht + 1024 * H_D, H_D, H_D); // rows 1024..2047
  build_head_vecs<<<8, 256, 0, stream>>>(Wc2, Wt2, Wk2, bc1, bt1, bk1, w2cat, bcat);

  // --- trunk GEMMs (block tile 64 x 256) ---
  gemm_relu_store_bf16<OBS_D, H_D><<<dim3(H_D / 256, B_ROWS / 64), 256, 0, stream>>>(
      obs_bf, W1t, b1, g1);
  gemm_relu_store_bf16<H_D, H_D><<<dim3(H_D / 256, B_ROWS / 64), 256, 0, stream>>>(
      g1, W2t, b2, g2);

  // --- fused heads GEMM + per-head dot partials ---
  heads_gemm_partial<H_D><<<dim3(2048 / 256, B_ROWS / 64), 256, 0, stream>>>(
      g2, Wht, bcat, w2cat, part);

  // --- finalize ---
  finalize_kernel<<<B_ROWS / 256, 256, 0, stream>>>(part, bc2, bt2, bk2, (float*)d_out);

  (void)in_sizes; (void)n_in; (void)out_size; (void)ws_size;
}
